// LocalHilbertAggregator_62079457296819
// MI455X (gfx1250) — matrix operand order repaired and checked by reading, not compiled, not run on hardware
//
#include <hip/hip_runtime.h>
#include <hip/hip_bf16.h>
#include <math.h>
#include <stdint.h>

#define DIMM     512
#define WIDTH_   640
#define HEIGHT_  480
#define D_STATE  16
#define D_CONV   4
#define HEADDIM  64
#define D_INNER  1024
#define NHEADS   16
#define CONV_DIM 1056
#define D_IN_PROJ 2096
#define NPAD     2176          // D_IN_PROJ padded to multiple of 128
#define BATCH    8
#define SEQ      4096
#define ROWS     (BATCH * SEQ) // 32768
#define EPSF     1e-5f

typedef __attribute__((ext_vector_type(16))) _Float16 v16h;
typedef __attribute__((ext_vector_type(8)))  _Float16 v8h;
typedef __attribute__((ext_vector_type(8)))  float    v8f;

__device__ __forceinline__ float siluf(float x) { return x / (1.f + expf(-x)); }

// CDNA5 async global->LDS copy (16B per lane), tracked by ASYNCcnt.
#define USE_ASYNC_LDS 1
__device__ __forceinline__ void async_copy16(void* lds, const void* g) {
#if USE_ASYNC_LDS
    asm volatile("global_load_async_to_lds_b128 %0, %1, off"
                 :: "v"((unsigned)(uintptr_t)lds),
                    "v"((unsigned long long)(uintptr_t)g)
                 : "memory");
#else
    *(v8h*)lds = *(const v8h*)g;
#endif
}
#if USE_ASYNC_LDS
#define WAIT_ASYNC(n) asm volatile("s_wait_asynccnt " #n ::: "memory")
#else
#define WAIT_ASYNC(n)
#endif

// ---------------------------------------------------------------- zero
__global__ void zero_kernel(float* p, size_t n) {
    size_t i = (size_t)blockIdx.x * blockDim.x + threadIdx.x;
    if (i < n) p[i] = 0.f;
}

// ---------------------------------------------------------------- stable argsort (bitonic on key*4096+idx)
__global__ __launch_bounds__(1024) void sort_kernel(const int* __restrict__ coords,
                                                    int* __restrict__ idx_out) {
    __shared__ unsigned s[SEQ];
    int b = blockIdx.x, ord = blockIdx.y;
    for (int i = threadIdx.x; i < SEQ; i += 1024) {
        int cx = coords[((size_t)b * SEQ + i) * 2 + 0];
        int cy = coords[((size_t)b * SEQ + i) * 2 + 1];
        unsigned key = (ord == 0) ? (unsigned)(cy * WIDTH_ + cx)
                                  : (unsigned)(cx * HEIGHT_ + cy);
        s[i] = key * 4096u + (unsigned)i;
    }
    __syncthreads();
    for (int k = 2; k <= SEQ; k <<= 1) {
        for (int j = k >> 1; j > 0; j >>= 1) {
            for (int i = threadIdx.x; i < SEQ; i += 1024) {
                int partner = i ^ j;
                if (partner > i) {
                    bool up = ((i & k) == 0);
                    unsigned a = s[i], c = s[partner];
                    if ((a > c) == up) { s[i] = c; s[partner] = a; }
                }
            }
            __syncthreads();
        }
    }
    for (int i = threadIdx.x; i < SEQ; i += 1024)
        idx_out[((size_t)(ord * BATCH + b)) * SEQ + i] = (int)(s[i] & 4095u);
}

// ---------------------------------------------------------------- weights -> f16 (W_in padded to NPAD cols)
__global__ void prep_weights(const float* __restrict__ W_in, const float* __restrict__ W_out,
                             _Float16* __restrict__ Wi16, _Float16* __restrict__ Wo16) {
    int i = blockIdx.x * blockDim.x + threadIdx.x;
    const int total1 = DIMM * NPAD;
    if (i < total1) {
        int rk = i / NPAD, c = i % NPAD;
        float v = (c < D_IN_PROJ) ? W_in[(size_t)rk * D_IN_PROJ + c] : 0.f;
        Wi16[i] = (_Float16)v;
    }
    int j = i - total1;
    if (j >= 0 && j < D_INNER * DIMM) Wo16[j] = (_Float16)W_out[j];
}

// ---------------------------------------------------------------- gather sorted rows -> f16
__global__ __launch_bounds__(256) void gather_kernel(const float* __restrict__ x,
                                                     const int* __restrict__ idx,
                                                     _Float16* __restrict__ u, int ord) {
    int r = blockIdx.x;
    int b = r >> 12, t = r & 4095;
    int src = idx[((size_t)(ord * BATCH + b)) * SEQ + t];
    const float* xr = x + ((size_t)(b * SEQ + src)) * DIMM;
    _Float16* ur = u + (size_t)r * DIMM;
    for (int d = threadIdx.x; d < DIMM; d += 256) ur[d] = (_Float16)xr[d];
}

// ---------------------------------------------------------------- WMMA GEMM: C(f16) = A(f16,MxK) * B(f16,KxN)
// 256 threads = 8 waves; block tile 128x128; wave tile 64x32 (4x2 wmma tiles);
// async global->LDS double-buffered pipeline; straight-line f16 epilogue (no spills).
__global__ __launch_bounds__(256) void gemm_wmma(
    const _Float16* __restrict__ A, const _Float16* __restrict__ Bw,
    int K, int lda, int ldb, _Float16* __restrict__ C, int ldc) {
    __shared__ _Float16 sA[2][128][40];   // 128x32 tile, padded stride
    __shared__ _Float16 sB[2][32][136];   // 32x128 tile, padded stride
    const int tid = threadIdx.x;
    const int mBase = blockIdx.x * 128;
    const int nBase = blockIdx.y * 128;
    const int lane = tid & 31, wv = tid >> 5;
    const int row = lane & 15, hi = lane >> 4;
    const int wm = (wv >> 2) * 64, wn = (wv & 3) * 32;

    v8f acc[4][2];
    for (int mt = 0; mt < 4; ++mt)
        for (int nt = 0; nt < 2; ++nt)
            for (int e = 0; e < 8; ++e) acc[mt][nt][e] = 0.f;

    const int arow = tid >> 1, aseg = (tid & 1) * 16;
    const int brow = tid >> 3, bseg = (tid & 7) * 16;

    const int steps = K >> 5;
    // prologue: issue stage 0 (4 async 16B copies per thread)
    {
        const _Float16* gA = A + (size_t)(mBase + arow) * lda + aseg;
        async_copy16(&sA[0][arow][aseg], gA);
        async_copy16(&sA[0][arow][aseg + 8], gA + 8);
        const _Float16* gB = Bw + (size_t)brow * ldb + nBase + bseg;
        async_copy16(&sB[0][brow][bseg], gB);
        async_copy16(&sB[0][brow][bseg + 8], gB + 8);
    }
    for (int s = 0; s < steps; ++s) {
        const int buf = s & 1;
        const bool more = (s + 1) < steps;
        if (more) {   // issue stage s+1 into other buffer
            int k0 = (s + 1) << 5;
            const _Float16* gA = A + (size_t)(mBase + arow) * lda + k0 + aseg;
            async_copy16(&sA[buf ^ 1][arow][aseg], gA);
            async_copy16(&sA[buf ^ 1][arow][aseg + 8], gA + 8);
            const _Float16* gB = Bw + (size_t)(k0 + brow) * ldb + nBase + bseg;
            async_copy16(&sB[buf ^ 1][brow][bseg], gB);
            async_copy16(&sB[buf ^ 1][brow][bseg + 8], gB + 8);
            WAIT_ASYNC(4);        // stage s complete (in-order), s+1 may fly
        } else {
            WAIT_ASYNC(0);
        }
        __syncthreads();

        v16h af[4], bf[2];
        for (int mt = 0; mt < 4; ++mt) {       // A frag: lane holds row M=row, K chunks per ISA layout
            int ar = wm + mt * 16 + row;
            v8h lo = *(const v8h*)&sA[buf][ar][hi * 8];
            v8h hh = *(const v8h*)&sA[buf][ar][16 + hi * 8];
            for (int e = 0; e < 8; ++e) { af[mt][e] = lo[e]; af[mt][8 + e] = hh[e]; }
        }
        // B frag: lane holds column N=row, K chunks per ISA layout (same as A)
        for (int nt = 0; nt < 2; ++nt) {
            int bc = wn + nt * 16 + row;
            for (int e = 0; e < 8; ++e) { bf[nt][e] = sB[buf][hi * 8 + e][bc]; bf[nt][8 + e] = sB[buf][16 + hi * 8 + e][bc]; }
        }
        for (int mt = 0; mt < 4; ++mt)
            for (int nt = 0; nt < 2; ++nt)
                acc[mt][nt] = __builtin_amdgcn_wmma_f32_16x16x32_f16(
                    false, af[mt], false, bf[nt], (short)0, acc[mt][nt], false, false);
        __syncthreads();
    }

    // straight-line epilogue: f16 stores, 16 lanes -> 32 contiguous bytes per row
    for (int mt = 0; mt < 4; ++mt)
        for (int nt = 0; nt < 2; ++nt) {
            int m0 = mBase + wm + mt * 16 + hi * 8;
            int n  = nBase + wn + nt * 16 + row;
            size_t base = (size_t)m0 * ldc + n;
            for (int e = 0; e < 8; ++e)
                C[base + (size_t)e * ldc] = (_Float16)acc[mt][nt][e];
        }
}

// ---------------------------------------------------------------- dt = softplus(raw+bias), dA = exp(-exp(A_log)*dt)
__global__ void dtda_kernel(const _Float16* __restrict__ zxb, const float* __restrict__ dt_bias,
                            const float* __restrict__ A_log, float* __restrict__ dtc,
                            float* __restrict__ dA) {
    int i = blockIdx.x * blockDim.x + threadIdx.x;
    if (i >= ROWS * NHEADS) return;
    int h = i & 15, r = i >> 4;
    float xv = (float)zxb[(size_t)r * NPAD + (D_INNER + CONV_DIM) + h] + dt_bias[h];
    float sp = (xv > 20.f) ? xv : log1pf(expf(xv));
    dtc[i] = sp;
    dA[i] = expf(-expf(A_log[h]) * sp);
}

// ---------------------------------------------------------------- causal depthwise conv (k=4) + SiLU
__global__ void conv_kernel(const _Float16* __restrict__ zxb, const float* __restrict__ conv_w,
                            const float* __restrict__ conv_b, _Float16* __restrict__ out) {
    size_t id = (size_t)blockIdx.x * blockDim.x + threadIdx.x;
    if (id >= (size_t)ROWS * CONV_DIM) return;
    int c = (int)(id % CONV_DIM);
    int r = (int)(id / CONV_DIM);
    int t = r & 4095;
    float acc = conv_b[c];
    for (int k = 0; k < D_CONV; ++k) {
        int tt = t + k - 3;
        if (tt >= 0)
            acc += (float)zxb[(size_t)(r + k - 3) * NPAD + D_INNER + c] * conv_w[k * CONV_DIM + c];
    }
    out[id] = (_Float16)siluf(acc);
}

// ---------------------------------------------------------------- selective scan; block = 64x16 state elements
__global__ __launch_bounds__(1024) void scan_kernel(const _Float16* __restrict__ cact,
                                                    const float* __restrict__ dA,
                                                    const float* __restrict__ dtc,
                                                    const float* __restrict__ D_ssm,
                                                    _Float16* __restrict__ y) {
    __shared__ _Float16 sx[64][64];
    __shared__ _Float16 sBm[64][16];
    __shared__ _Float16 sCm[64][16];
    __shared__ float sdA[64], sdt[64];
    const int head = blockIdx.x, b = blockIdx.y;
    const int tid = threadIdx.x;
    const int p = tid >> 4, n = tid & 15;
    const float Dh = D_ssm[head];
    float h = 0.f;
    for (int ch = 0; ch < SEQ / 64; ++ch) {
        int r0 = b * SEQ + ch * 64;
        for (int i = tid; i < 64 * 64; i += 1024) {
            int rr = i >> 6, cc = i & 63;
            sx[rr][cc] = cact[(size_t)(r0 + rr) * CONV_DIM + head * HEADDIM + cc];
        }
        {
            int rr = tid >> 4, cc = tid & 15;
            sBm[rr][cc] = cact[(size_t)(r0 + rr) * CONV_DIM + D_INNER + cc];
            sCm[rr][cc] = cact[(size_t)(r0 + rr) * CONV_DIM + D_INNER + D_STATE + cc];
        }
        if (tid < 64) {
            sdA[tid] = dA[(size_t)(r0 + tid) * NHEADS + head];
            sdt[tid] = dtc[(size_t)(r0 + tid) * NHEADS + head];
        }
        __syncthreads();
        for (int i = 0; i < 64; ++i) {
            float xv = (float)sx[i][p];
            h = h * sdA[i] + xv * sdt[i] * (float)sBm[i][n];
            float c = h * (float)sCm[i][n];
            c += __shfl_xor(c, 1, 16);
            c += __shfl_xor(c, 2, 16);
            c += __shfl_xor(c, 4, 16);
            c += __shfl_xor(c, 8, 16);
            if (n == 0)
                y[(size_t)(r0 + i) * D_INNER + head * HEADDIM + p] = (_Float16)(c + Dh * xv);
        }
        __syncthreads();
    }
}

// ---------------------------------------------------------------- y = rmsnorm(y * silu(z)) * rms_w (in place, f16)
__global__ __launch_bounds__(256) void gate_norm_kernel(_Float16* __restrict__ y,
                                                        const _Float16* __restrict__ zxb,
                                                        const float* __restrict__ rms_w) {
    __shared__ float red[256];
    int r = blockIdx.x, tid = threadIdx.x;
    float v[4], ss = 0.f;
    for (int i = 0; i < 4; ++i) {
        int d = tid + i * 256;
        float zv = (float)zxb[(size_t)r * NPAD + d];
        float vv = (float)y[(size_t)r * D_INNER + d] * siluf(zv);
        v[i] = vv; ss += vv * vv;
    }
    red[tid] = ss; __syncthreads();
    for (int off = 128; off > 0; off >>= 1) { if (tid < off) red[tid] += red[tid + off]; __syncthreads(); }
    float scale = rsqrtf(red[0] / (float)D_INNER + EPSF);
    for (int i = 0; i < 4; ++i) {
        int d = tid + i * 256;
        y[(size_t)r * D_INNER + d] = (_Float16)(v[i] * scale * rms_w[d]);
    }
}

// ---------------------------------------------------------------- scatter back to original order, accumulate *0.5
__global__ __launch_bounds__(256) void scatter_add_kernel(const _Float16* __restrict__ o16,
                                                          const int* __restrict__ idx,
                                                          float* __restrict__ accum, int ord) {
    int r = blockIdx.x;
    int b = r >> 12, t = r & 4095;
    int dst = idx[((size_t)(ord * BATCH + b)) * SEQ + t];
    const _Float16* src = o16 + (size_t)r * DIMM;
    float* dp = accum + ((size_t)(b * SEQ + dst)) * DIMM;
    for (int d = threadIdx.x; d < DIMM; d += 256) dp[d] += 0.5f * (float)src[d];
}

// ---------------------------------------------------------------- seg = (mean+max)/2 over L, then LayerNorm over DIM
__global__ __launch_bounds__(512) void segln_kernel(const float* __restrict__ accum,
                                                    const float* __restrict__ ln_g,
                                                    const float* __restrict__ ln_b,
                                                    float* __restrict__ out) {
    __shared__ float red[512];
    int b = blockIdx.x, d = threadIdx.x;
    float sum = 0.f, mx = -3.4e38f;
    for (int t = 0; t < SEQ; ++t) {
        float v = accum[((size_t)(b * SEQ + t)) * DIMM + d];
        sum += v; mx = fmaxf(mx, v);
    }
    float seg = (sum / (float)SEQ + mx) * 0.5f;
    red[d] = seg; __syncthreads();
    for (int off = 256; off > 0; off >>= 1) { if (d < off) red[d] += red[d + off]; __syncthreads(); }
    float mu = red[0] / (float)DIMM; __syncthreads();
    float df = seg - mu;
    red[d] = df * df; __syncthreads();
    for (int off = 256; off > 0; off >>= 1) { if (d < off) red[d] += red[d + off]; __syncthreads(); }
    float var = red[0] / (float)DIMM;
    out[(size_t)b * DIMM + d] = df * rsqrtf(var + EPSF) * ln_g[d] + ln_b[d];
}

// ================================================================ host
extern "C" void kernel_launch(void* const* d_in, const int* in_sizes, int n_in,
                              void* d_out, int out_size, void* d_ws, size_t ws_size,
                              hipStream_t stream) {
    (void)in_sizes; (void)n_in; (void)out_size; (void)ws_size;
    const float* x       = (const float*)d_in[0];
    const int*   coords  = (const int*)  d_in[1];
    const float* W_in    = (const float*)d_in[2];
    const float* conv_w  = (const float*)d_in[3];
    const float* conv_b  = (const float*)d_in[4];
    const float* dt_bias = (const float*)d_in[5];
    const float* A_log   = (const float*)d_in[6];
    const float* D_ssm   = (const float*)d_in[7];
    const float* rms_w   = (const float*)d_in[8];
    const float* W_out   = (const float*)d_in[9];
    const float* ln_g    = (const float*)d_in[10];
    const float* ln_b    = (const float*)d_in[11];
    float* out = (float*)d_out;

    char* ws = (char*)d_ws;
    size_t off = 0;
    auto alloc = [&](size_t bytes) -> void* {
        void* p = ws + off;
        off = (off + bytes + 255) & ~(size_t)255;
        return p;
    };
    float*    accum = (float*)   alloc((size_t)ROWS * DIMM * 4);
    int*      idxb  = (int*)     alloc((size_t)2 * ROWS * 4);
    _Float16* Wi16  = (_Float16*)alloc((size_t)DIMM * NPAD * 2);
    _Float16* Wo16  = (_Float16*)alloc((size_t)D_INNER * DIMM * 2);
    _Float16* u16   = (_Float16*)alloc((size_t)ROWS * DIMM * 2);
    _Float16* zxb16 = (_Float16*)alloc((size_t)ROWS * NPAD * 2);   // z | xBC | dt (f16, stride NPAD)
    float*    dtc   = (float*)   alloc((size_t)ROWS * NHEADS * 4);
    float*    dAb   = (float*)   alloc((size_t)ROWS * NHEADS * 4);
    _Float16* cact  = (_Float16*)alloc((size_t)ROWS * CONV_DIM * 2);
    _Float16* ybuf  = (_Float16*)alloc((size_t)ROWS * D_INNER * 2);
    _Float16* o16   = (_Float16*)alloc((size_t)ROWS * DIMM * 2);

    zero_kernel<<<(unsigned)(((size_t)ROWS * DIMM + 255) / 256), 256, 0, stream>>>(accum, (size_t)ROWS * DIMM);
    sort_kernel<<<dim3(BATCH, 2), 1024, 0, stream>>>(coords, idxb);
    {
        int total = DIMM * NPAD + D_INNER * DIMM;
        prep_weights<<<(total + 255) / 256, 256, 0, stream>>>(W_in, W_out, Wi16, Wo16);
    }
    for (int ord = 0; ord < 2; ++ord) {
        gather_kernel<<<ROWS, 256, 0, stream>>>(x, idxb, u16, ord);
        gemm_wmma<<<dim3(ROWS / 128, NPAD / 128), 256, 0, stream>>>(
            u16, Wi16, DIMM, DIMM, NPAD, zxb16, NPAD);
        dtda_kernel<<<(ROWS * NHEADS + 255) / 256, 256, 0, stream>>>(zxb16, dt_bias, A_log, dtc, dAb);
        {
            size_t total = (size_t)ROWS * CONV_DIM;
            conv_kernel<<<(unsigned)((total + 255) / 256), 256, 0, stream>>>(zxb16, conv_w, conv_b, cact);
        }
        scan_kernel<<<dim3(NHEADS, BATCH), 1024, 0, stream>>>(cact, dAb, dtc, D_ssm, ybuf);
        gate_norm_kernel<<<ROWS, 256, 0, stream>>>(ybuf, zxb16, rms_w);
        gemm_wmma<<<dim3(ROWS / 128, DIMM / 128), 256, 0, stream>>>(
            ybuf, Wo16, D_INNER, D_INNER, DIMM, o16, DIMM);
        scatter_add_kernel<<<ROWS, 256, 0, stream>>>(o16, idxb, accum, ord);
    }
    segln_kernel<<<BATCH, 512, 0, stream>>>(accum, ln_g, ln_b, out);
}
